// VectorQuantizer_2911987827386
// MI455X (gfx1250) — matrix operand order repaired and checked by reading, not compiled, not run on hardware
//
#include <hip/hip_runtime.h>

// ---------------------------------------------------------------------------
// VQ-VAE vector quantizer for MI455X (gfx1250, wave32, WMMA + async-to-LDS).
//   x:        [32, 256, 64, 64] f32
//   codebook: [1024, 256] f32
// Outputs (concatenated in d_out as f32):
//   z_q_out [32,256,64,64], idx [131072] (as float), loss [1]
// ---------------------------------------------------------------------------

typedef __attribute__((ext_vector_type(16))) __bf16 v16bf;
typedef __attribute__((ext_vector_type(8)))  float  v8f;

#define K_CODES 1024
#define DIM     256
#define NVEC    (32 * 64 * 64)                 // 131072 vectors
#define NELEM   ((size_t)32 * 256 * 64 * 64)   // 33,554,432 elements
#define MTILE   128                            // z rows per block
#define ROWP    264                            // padded LDS row stride (bf16), 16B aligned
#define GATHER_BLOCKS 8192

// CDNA5 async copy: cache/memory -> LDS, tracked by ASYNCcnt (inline asm per
// bridge-doc guidance; VDST VGPR carries the per-lane LDS byte offset).
#define ASYNC_LOAD_B128(ldsOff, gptr)                                   \
  asm volatile("global_load_async_to_lds_b128 %0, %1, off"              \
               :: "v"(ldsOff), "v"(gptr) : "memory")
#define WAIT_ASYNCCNT(n)                                                \
  asm volatile("s_wait_asynccnt " #n ::: "memory")

// ---- Kernel A: codebook -> bf16 copy + squared norms ---------------------
__global__ __launch_bounds__(256) void prep_codebook(const float* __restrict__ cb,
                                                     __bf16* __restrict__ cbb,
                                                     float* __restrict__ enorm) {
  const int k = blockIdx.x;     // one code per block
  const int t = threadIdx.x;    // 256 threads == DIM
  float v = cb[(size_t)k * DIM + t];
  cbb[(size_t)k * DIM + t] = (__bf16)v;
  float sq = v * v;
#pragma unroll
  for (int off = 16; off > 0; off >>= 1) sq += __shfl_down(sq, off, 32);
  __shared__ float red[8];
  if ((t & 31) == 0) red[t >> 5] = sq;
  __syncthreads();
  if (t == 0) {
    float s = 0.f;
#pragma unroll
    for (int i = 0; i < 8; ++i) s += red[i];
    enorm[k] = s;
  }
}

// ---- Kernel B: transpose x [b,c,h,w] -> z_bf16 [n][c], n=(b*64+h)*64+w ---
__global__ __launch_bounds__(256) void transpose_to_bf16(const float* __restrict__ x,
                                                         __bf16* __restrict__ zt) {
  __shared__ float tile[32][33];
  const int n0 = blockIdx.x * 32;   // 32 consecutive n share (b,h)
  const int c0 = blockIdx.y * 32;
  const int tx = threadIdx.x;       // 0..31
  const int ty = threadIdx.y;       // 0..7
  const int b  = n0 >> 12;
  const int h  = (n0 >> 6) & 63;
  const int w0 = n0 & 63;
  const float* xp = x + ((size_t)b * DIM * 64 + h) * 64 + w0 + tx;
#pragma unroll
  for (int j = 0; j < 4; ++j) {
    int c = c0 + ty + 8 * j;
    tile[ty + 8 * j][tx] = xp[(size_t)c * 4096];   // coalesced along w
  }
  __syncthreads();
#pragma unroll
  for (int j = 0; j < 4; ++j) {
    int n = n0 + ty + 8 * j;
    zt[(size_t)n * DIM + c0 + tx] = (__bf16)tile[tx][ty + 8 * j];  // coalesced along c
  }
}

// ---- Kernel C: WMMA distance GEMM + fused argmin -------------------------
// dist(n,m) = ||e_m||^2 - 2 * z_n . e_m   (||z||^2 constant per row: dropped)
// A = codebook tile (16 codes x 32 dims), B = z rows (held in VGPRs for the
// whole sweep), double-buffered codebook chunks fed by async-to-LDS copies.
__global__ __launch_bounds__(256) void vq_argmin(const __bf16* __restrict__ zt,
                                                 const __bf16* __restrict__ cbb,
                                                 const float* __restrict__ enorm,
                                                 int* __restrict__ idx_out) {
  __shared__ __bf16 buf0[128 * ROWP];   // 67,584 B: z tile first, then chunks 0,2,4,6
  __shared__ __bf16 buf1[128 * ROWP];   // 67,584 B: chunks 1,3,5,7
  __shared__ float  en_s[K_CODES];      //  4,096 B

  const int tid  = threadIdx.x;
  const int lane = tid & 31;
  const int wave = tid >> 5;
  const int hi   = lane >> 4;           // 0: lanes 0-15, 1: lanes 16-31
  const int lm   = lane & 15;
  const int rowBase = blockIdx.x * MTILE;
  const int myRow   = wave * 16 + lm;   // B-fragment column == z row in tile

  // --- Prologue: stage z tile into buf0 (coalesced), hoist B frags to VGPRs.
  {
    const uint4* src = (const uint4*)(zt + (size_t)rowBase * DIM);
    for (int i = tid; i < MTILE * 32; i += 256) {
      int r = i >> 5, p = i & 31;
      *(uint4*)(buf0 + r * ROWP + p * 8) = src[r * 32 + p];
    }
  }
  for (int i = tid; i < K_CODES; i += 256) en_s[i] = enorm[i];
  __syncthreads();

  union Frag { v16bf v; uint4 q[2]; };
  Frag bfr[8];                          // 64 VGPRs: z row, all 256 dims
  {
    const __bf16* zrow = buf0 + myRow * ROWP;
#pragma unroll
    for (int kk = 0; kk < 8; ++kk) {
      // B (z, 32x16): low lanes K k0..k0+15, high lanes k0+16..k0+31
      bfr[kk].q[0] = *(const uint4*)(zrow + kk * 32 + 16 * hi);
      bfr[kk].q[1] = *(const uint4*)(zrow + kk * 32 + 16 * hi + 8);
    }
  }
  __syncthreads();                      // all frag reads done before buf0 reuse

  // --- Issue async staging for chunks 0 and 1 (16 async ops per wave each).
  // Each chunk: 128 codebook rows x 32 uint4, spread over 256 threads.
  const unsigned b0off = (unsigned)(size_t)(buf0);
  const unsigned b1off = (unsigned)(size_t)(buf1);
#pragma unroll 4
  for (int i = 0; i < 16; ++i) {
    int e = tid + i * 256;              // 0..4095
    int r = e >> 5, p = e & 31;
    unsigned lo = (unsigned)(r * ROWP * 2 + p * 16);
    const uint4* g = (const uint4*)cbb + (size_t)r * 32 + p;
    ASYNC_LOAD_B128(b0off + lo, g);
    ASYNC_LOAD_B128(b1off + lo, g + 128 * 32);
  }

  float bestVal = 3.4e38f;
  int   bestIdx = 0;

  for (int c = 0; c < 8; ++c) {         // 8 chunks of 128 codes
    const __bf16* cur = (c & 1) ? buf1 : buf0;
    // Async loads complete in order; each wave issued 16 per chunk, so
    // <=16 outstanding means chunk c fully landed for c >= 1; at c == 0 (prologue interleaves chunks 0/1) only its first 8 loads are sure.
    if (c == 7) { WAIT_ASYNCCNT(0); } else { WAIT_ASYNCCNT(16); }
    __syncthreads();

    for (int sc = 0; sc < 8; ++sc) {    // 16-code sub-tiles
      const __bf16* crow = cur + (sc * 16 + lm) * ROWP;
      v8f acc = {0.f, 0.f, 0.f, 0.f, 0.f, 0.f, 0.f, 0.f};
#pragma unroll
      for (int kk = 0; kk < 8; ++kk) {
        int k0 = kk * 32;
        Frag a;
        // A (codebook 16x32): low lanes K k0..k0+15, high lanes k0+16..k0+31
        a.q[0] = *(const uint4*)(crow + k0 + 16 * hi);
        a.q[1] = *(const uint4*)(crow + k0 + 16 * hi + 8);
        acc = __builtin_amdgcn_wmma_f32_16x16x32_bf16(false, a.v, false, bfr[kk].v,
                                                      (short)0, acc, false, false);
      }
      // Lane holds codes m = v + 8*hi of this sub-tile, for z row `myRow`.
#pragma unroll
      for (int v = 0; v < 8; ++v) {
        int m = c * 128 + sc * 16 + 8 * hi + v;
        float d = en_s[m] - 2.0f * acc[v];
        if (d < bestVal) { bestVal = d; bestIdx = m; }  // strict < keeps smallest idx
      }
    }

    __syncthreads();                    // everyone done reading `cur`
    if (c + 2 < 8) {                    // refill this buffer with chunk c+2
      const unsigned boff = (c & 1) ? b1off : b0off;
      const uint4* gbase = (const uint4*)cbb + (size_t)(c + 2) * 128 * 32;
#pragma unroll 4
      for (int i = 0; i < 16; ++i) {
        int e = tid + i * 256;
        int r = e >> 5, p = e & 31;
        ASYNC_LOAD_B128(boff + (unsigned)(r * ROWP * 2 + p * 16), gbase + r * 32 + p);
      }
    }
  }

  // Fold lane pair (lm, lm+16); on ties take smaller code index (jnp.argmin).
  float oVal = __shfl_xor(bestVal, 16, 32);
  int   oIdx = __shfl_xor(bestIdx, 16, 32);
  if (oVal < bestVal || (oVal == bestVal && oIdx < bestIdx)) { bestVal = oVal; bestIdx = oIdx; }
  if (hi == 0) idx_out[rowBase + myRow] = bestIdx;
}

// ---- Kernel D: gather z_q (exact f32), write [b,c,h,w], loss partials ----
__global__ __launch_bounds__(256) void gather_out(const float* __restrict__ x,
                                                  const float* __restrict__ cb,
                                                  const int* __restrict__ idx,
                                                  float* __restrict__ zq_out,
                                                  double* __restrict__ lossPart) {
  double lsum = 0.0;
  for (size_t o = (size_t)blockIdx.x * 256 + threadIdx.x; o < NELEM;
       o += (size_t)gridDim.x * 256) {
    int w = (int)(o & 63);
    int h = (int)((o >> 6) & 63);
    int c = (int)((o >> 12) & 255);
    int b = (int)(o >> 20);
    int n = (b << 12) | (h << 6) | w;
    float zq = cb[(size_t)idx[n] * DIM + c];   // exact f32 gather (L2-resident)
    float d  = x[o] - zq;
    lsum += (double)(d * d);
    zq_out[o] = zq;                            // straight-through forward value
  }
  __shared__ double red[256];
  red[threadIdx.x] = lsum;
  __syncthreads();
  for (int s = 128; s > 0; s >>= 1) {
    if (threadIdx.x < s) red[threadIdx.x] += red[threadIdx.x + s];
    __syncthreads();
  }
  if (threadIdx.x == 0) lossPart[blockIdx.x] = red[0];
}

// ---- Kernel E1: deterministic loss reduction -----------------------------
__global__ __launch_bounds__(256) void finalize_loss(const double* __restrict__ part,
                                                     float* __restrict__ outLoss) {
  __shared__ double red[256];
  double s = 0.0;
  for (int i = threadIdx.x; i < GATHER_BLOCKS; i += 256) s += part[i];
  red[threadIdx.x] = s;
  __syncthreads();
  for (int st = 128; st > 0; st >>= 1) {
    if (threadIdx.x < st) red[threadIdx.x] += red[threadIdx.x + st];
    __syncthreads();
  }
  if (threadIdx.x == 0)
    outLoss[0] = (float)(2.0 * red[0] / (double)NELEM);  // both loss terms equal in fwd
}

// ---- Kernel E2: idx -> float output --------------------------------------
__global__ __launch_bounds__(256) void write_idx(const int* __restrict__ idx,
                                                 float* __restrict__ outIdx) {
  int i = blockIdx.x * 256 + threadIdx.x;
  if (i < NVEC) outIdx[i] = (float)idx[i];
}

// ---------------------------------------------------------------------------
extern "C" void kernel_launch(void* const* d_in, const int* in_sizes, int n_in,
                              void* d_out, int out_size, void* d_ws, size_t ws_size,
                              hipStream_t stream) {
  (void)in_sizes; (void)n_in; (void)out_size; (void)ws_size;
  const float* x  = (const float*)d_in[0];
  const float* cb = (const float*)d_in[1];

  // Workspace layout (~68.2 MB).
  char* ws = (char*)d_ws;
  __bf16* zt    = (__bf16*)ws;                                       // 67,108,864 B
  __bf16* cbb   = (__bf16*)(ws + 67108864);                          //    524,288 B
  float*  enorm = (float*) (ws + 67108864 + 524288);                 //      4,096 B
  int*    idx   = (int*)   (ws + 67108864 + 524288 + 4096);          //    524,288 B
  double* lossP = (double*)(ws + 67108864 + 524288 + 4096 + 524288); //     65,536 B

  float* zq_out  = (float*)d_out;
  float* outIdx  = zq_out + NELEM;
  float* outLoss = outIdx + NVEC;

  prep_codebook    <<<K_CODES, 256, 0, stream>>>(cb, cbb, enorm);
  transpose_to_bf16<<<dim3(NVEC / 32, DIM / 32), dim3(32, 8), 0, stream>>>(x, zt);
  vq_argmin        <<<NVEC / MTILE, 256, 0, stream>>>(zt, cbb, enorm, idx);
  gather_out       <<<GATHER_BLOCKS, 256, 0, stream>>>(x, cb, idx, zq_out, lossP);
  finalize_loss    <<<1, 256, 0, stream>>>(lossP, outLoss);
  write_idx        <<<(NVEC + 255) / 256, 256, 0, stream>>>(idx, outIdx);
}
